// FFB_encoder_63943473103243
// MI455X (gfx1250) — matrix operand order repaired and checked by reading, not compiled, not run on hardware
//
#include <hip/hip_runtime.h>

// ---- architecture constants (match reference) ----
#define N_PTS   262144
#define DIM     256
#define NLVL    6
#define TSIZE   524288          // hash table entries per level (2^19)
#define TILE_M  128             // points per workgroup
#define XSTR    264             // padded LDS stride (halfs) for activations
#define WSTR    264             // padded LDS stride (halfs) for weights
#define TWO_PI  6.28318530717958647692f
#define W0SIN   5.0f            // L-1
#define W0HIGH  10.0f           // 2*(L-1)

typedef __attribute__((ext_vector_type(16))) _Float16 v16h;
typedef __attribute__((ext_vector_type(8)))  _Float16 v8h;
typedef __attribute__((ext_vector_type(8)))  float    v8f;

// ---- dynamic LDS layout (bytes) ----
#define OFF_W    0
#define OFF_XA   (OFF_W   + 256*WSTR*2)        // 135168
#define OFF_XB   (OFF_XA  + TILE_M*XSTR*2)     // 202752
#define OFF_ASC  (OFF_XB  + TILE_M*XSTR*2)     // 270336
#define OFF_BSS  (OFF_ASC + NLVL*2*DIM*4)      // 282624
#define OFF_BHS  (OFF_BSS + NLVL*DIM*4)        // 288768
#define OFF_G    (OFF_BHS + NLVL*DIM*4)        // 294912
#define OFF_POS  (OFF_G   + NLVL*TILE_M*2*4)   // 301056
#define SHMEM_BYTES (OFF_POS + TILE_M*3*4)     // 302592  (< 320KB)

// Convert Ws/Wh (f32, [L][k][n]) -> f16 transposed [L][n][k] so that a WMMA
// B-fragment (column n of W) is row-contiguous in LDS.
__global__ void ffb_prep_weights(const float* __restrict__ Ws,
                                 const float* __restrict__ Wh,
                                 _Float16* __restrict__ wsT,
                                 _Float16* __restrict__ whT) {
    int idx = blockIdx.x * blockDim.x + threadIdx.x;
    const int per = NLVL * DIM * DIM;          // 393216
    const float* src; _Float16* dst; int e;
    if (idx < per) { src = Ws; dst = wsT; e = idx; }
    else           { src = Wh; dst = whT; e = idx - per; }
    int l = e >> 16;           // level
    int r = e & 65535;
    int n = r >> 8;            // output column
    int k = r & 255;           // input row
    dst[e] = (_Float16)src[(l << 16) + (k << 8) + n];
}

__device__ __forceinline__ void stage_weights(const _Float16* __restrict__ g,
                                              _Float16* Wbuf, int tid) {
    // 256x256 halfs = 8192 b128 chunks, perfectly coalesced.
    #pragma unroll
    for (int it = 0; it < 32; ++it) {
        int c  = tid + it * 256;
        int rw = c >> 5;        // row (output column n)
        int c8 = c & 31;        // 8-half chunk within row
        *(v8h*)(Wbuf + rw * WSTR + c8 * 8) = *(const v8h*)(g + c * 8);
    }
}

__device__ __forceinline__ v16h cat16(v8h lo, v8h hi) {
    return __builtin_shufflevector(lo, hi, 0,1,2,3,4,5,6,7,8,9,10,11,12,13,14,15);
}

__global__ __launch_bounds__(256, 1)
void ffb_fused(const float* __restrict__ in_pos,
               const float* __restrict__ table,
               const float* __restrict__ ffn_A,
               const float* __restrict__ W0,
               const float* __restrict__ b0,
               const float* __restrict__ bs,
               const float* __restrict__ bh,
               const _Float16* __restrict__ wsT,
               const _Float16* __restrict__ whT,
               float* __restrict__ out) {
    extern __shared__ char smem[];
    _Float16* Wbuf = (_Float16*)(smem + OFF_W);
    _Float16* Xa   = (_Float16*)(smem + OFF_XA);
    _Float16* Xb   = (_Float16*)(smem + OFF_XB);
    float*    Asc  = (float*)(smem + OFF_ASC);
    float*    bss  = (float*)(smem + OFF_BSS);
    float*    bhs  = (float*)(smem + OFF_BHS);
    float*    gfe  = (float*)(smem + OFF_G);
    float*    poss = (float*)(smem + OFF_POS);

    const int tid   = threadIdx.x;
    const int lane  = tid & 31;
    const int wave  = tid >> 5;
    const int row0  = blockIdx.x * TILE_M;
    const int wm0   = wave * 16;           // this wave's local row base
    const int lan15 = lane & 15;
    const int khalf = lane >> 4;
    const int mbase = wm0 + (khalf << 3);  // C-fragment row base per lane half

    // ---- stage positions ----
    if (tid < TILE_M) {
        size_t gi = (size_t)(row0 + tid) * 3;
        poss[tid * 3 + 0] = in_pos[gi + 0];
        poss[tid * 3 + 1] = in_pos[gi + 1];
        poss[tid * 3 + 2] = in_pos[gi + 2];
    }
    // ---- per-column constants: 2*pi*sigma_l*ffn_A, biases ----
    {
        int n = tid;
        #pragma unroll
        for (int l = 0; l < NLVL; ++l) {
            float sig = TWO_PI * (5.0f * (float)(1 << l));
            Asc[(l * 2 + 0) * DIM + n] = sig * ffn_A[(l * 2 + 0) * DIM + n];
            Asc[(l * 2 + 1) * DIM + n] = sig * ffn_A[(l * 2 + 1) * DIM + n];
            bss[l * DIM + n] = bs[l * DIM + n];
            bhs[l * DIM + n] = bh[l * DIM + n];
        }
    }
    __syncthreads();

    // ---- multi-res hash grid: g[l][m][2] ----
    if (tid < TILE_M) {
        float px = poss[tid * 3 + 0], py = poss[tid * 3 + 1], pz = poss[tid * 3 + 2];
        float ux = (px + 1.f) * 0.5f, uy = (py + 1.f) * 0.5f, uz = (pz + 1.f) * 0.5f;
        #pragma unroll 1
        for (int l = 0; l < NLVL; ++l) {
            float res = (float)(16 << l);
            float sx = ux * res, sy = uy * res, sz = uz * res;
            float flx = floorf(sx), fly = floorf(sy), flz = floorf(sz);
            float frx = sx - flx, fry = sy - fly, frz = sz - flz;
            unsigned cx = (unsigned)(int)flx, cy = (unsigned)(int)fly, cz = (unsigned)(int)flz;
            const float* tl = table + (size_t)l * TSIZE * 2;
            float g0 = 0.f, g1 = 0.f;
            #pragma unroll
            for (int c = 0; c < 8; ++c) {
                unsigned ox = (c >> 2) & 1u, oy = (c >> 1) & 1u, oz = c & 1u;
                unsigned h = ((cx + ox) * 1u) ^ ((cy + oy) * 2654435761u) ^ ((cz + oz) * 805459861u);
                h &= (TSIZE - 1u);
                float w = (ox ? frx : 1.f - frx) * (oy ? fry : 1.f - fry) * (oz ? frz : 1.f - frz);
                g0 += w * tl[(size_t)h * 2 + 0];
                g1 += w * tl[(size_t)h * 2 + 1];
            }
            gfe[(l * TILE_M + tid) * 2 + 0] = g0;
            gfe[(l * TILE_M + tid) * 2 + 1] = g1;
        }
    }

    // ---- layer 0 SIREN (K=3, scalar): each thread owns column n=tid ----
    {
        int n = tid;
        float w0 = W0[n], w1 = W0[DIM + n], w2 = W0[2 * DIM + n], bb = b0[n];
        #pragma unroll 4
        for (int m = 0; m < TILE_M; ++m) {
            float s = poss[m * 3 + 0] * w0 + poss[m * 3 + 1] * w1 + poss[m * 3 + 2] * w2 + bb;
            Xa[m * XSTR + n] = (_Float16)__sinf(W0SIN * s);
        }
    }

    v8f xout[16];
    #pragma unroll
    for (int j = 0; j < 16; ++j) xout[j] = (v8f)(0.f);

    _Float16* Xcur = Xa;
    _Float16* Xnxt = Xb;

    #pragma unroll 1
    for (int i = 0; i < NLVL; ++i) {
        __syncthreads();                      // previous readers of Wbuf done
        stage_weights(wsT + (size_t)i * DIM * DIM, Wbuf, tid);
        __syncthreads();                      // Wbuf + Xcur ready

        // A fragments for this wave (reused across all 16 N-blocks)
        const _Float16* xrow = Xcur + (wm0 + lan15) * XSTR;
        v16h afr[8];
        #pragma unroll
        for (int kb = 0; kb < 8; ++kb) {
            const _Float16* p = xrow + kb * 32 + khalf * 8;
            afr[kb] = cat16(*(const v8h*)(p), *(const v8h*)(p + 16));
        }

        // ---- GEMM 1: Xnxt = sin(5*(x@Ws + bs)) + sin(grid fourier) ----
        #pragma unroll
        for (int j = 0; j < 16; ++j) {
            const _Float16* wr = Wbuf + (j * 16 + lan15) * WSTR + khalf * 8;
            v8f acc = (v8f)(0.f);
            v8h lo = *(const v8h*)(wr);
            v8h hh = *(const v8h*)(wr + 16);
            #pragma unroll
            for (int kb = 0; kb < 8; ++kb) {
                v16h b = cat16(lo, hh);
                if (kb < 7) {                  // prefetch next B fragment
                    lo = *(const v8h*)(wr + (kb + 1) * 32);
                    hh = *(const v8h*)(wr + (kb + 1) * 32 + 16);
                }
                acc = __builtin_amdgcn_wmma_f32_16x16x32_f16(false, afr[kb], false, b,
                                                             (short)0, acc, false, false);
            }
            int n = j * 16 + lan15;
            float a0  = Asc[(i * 2 + 0) * DIM + n];
            float a1  = Asc[(i * 2 + 1) * DIM + n];
            float bsn = bss[i * DIM + n];
            #pragma unroll
            for (int r = 0; r < 8; ++r) {
                int m = mbase + r;
                float g0 = gfe[(i * TILE_M + m) * 2 + 0];
                float g1 = gfe[(i * TILE_M + m) * 2 + 1];
                float v  = __sinf(W0SIN * (acc[r] + bsn)) + __sinf(g0 * a0 + g1 * a1);
                Xnxt[m * XSTR + n] = (_Float16)v;   // other buffer: no barrier needed
            }
        }
        __syncthreads();                      // Wbuf free, Xnxt fully written
        stage_weights(whT + (size_t)i * DIM * DIM, Wbuf, tid);
        __syncthreads();                      // Wh + Xnxt ready

        // A fragments from the new activations
        const _Float16* xrow2 = Xnxt + (wm0 + lan15) * XSTR;
        #pragma unroll
        for (int kb = 0; kb < 8; ++kb) {
            const _Float16* p = xrow2 + kb * 32 + khalf * 8;
            afr[kb] = cat16(*(const v8h*)(p), *(const v8h*)(p + 16));
        }

        // ---- GEMM 2: xout += sin(10*(x@Wh + bh)) ----
        #pragma unroll
        for (int j = 0; j < 16; ++j) {
            const _Float16* wr = Wbuf + (j * 16 + lan15) * WSTR + khalf * 8;
            v8f acc = (v8f)(0.f);
            v8h lo = *(const v8h*)(wr);
            v8h hh = *(const v8h*)(wr + 16);
            #pragma unroll
            for (int kb = 0; kb < 8; ++kb) {
                v16h b = cat16(lo, hh);
                if (kb < 7) {
                    lo = *(const v8h*)(wr + (kb + 1) * 32);
                    hh = *(const v8h*)(wr + (kb + 1) * 32 + 16);
                }
                acc = __builtin_amdgcn_wmma_f32_16x16x32_f16(false, afr[kb], false, b,
                                                             (short)0, acc, false, false);
            }
            int n = j * 16 + lan15;
            float bhn = bhs[i * DIM + n];
            #pragma unroll
            for (int r = 0; r < 8; ++r)
                xout[j][r] += __sinf(W0HIGH * (acc[r] + bhn));
        }

        // swap activation buffers for next layer
        _Float16* t = Xcur; Xcur = Xnxt; Xnxt = t;
    }

    // ---- final store: [N,256] f32, 16 contiguous floats per lane-group ----
    #pragma unroll
    for (int j = 0; j < 16; ++j) {
        int n = j * 16 + lan15;
        #pragma unroll
        for (int r = 0; r < 8; ++r)
            out[(size_t)(row0 + mbase + r) * DIM + n] = xout[j][r];
    }
}

extern "C" void kernel_launch(void* const* d_in, const int* in_sizes, int n_in,
                              void* d_out, int out_size, void* d_ws, size_t ws_size,
                              hipStream_t stream) {
    (void)in_sizes; (void)n_in; (void)out_size; (void)ws_size;
    const float* in_pos = (const float*)d_in[0];
    const float* table  = (const float*)d_in[1];
    const float* ffn_A  = (const float*)d_in[2];
    const float* W0     = (const float*)d_in[3];
    const float* b0     = (const float*)d_in[4];
    const float* Ws     = (const float*)d_in[5];
    const float* bs     = (const float*)d_in[6];
    const float* Wh     = (const float*)d_in[7];
    const float* bh     = (const float*)d_in[8];
    float* out = (float*)d_out;

    _Float16* wsT = (_Float16*)d_ws;
    _Float16* whT = wsT + (size_t)NLVL * DIM * DIM;

    ffb_prep_weights<<<(2 * NLVL * DIM * DIM) / 256, 256, 0, stream>>>(Ws, Wh, wsT, whT);

    hipFuncSetAttribute((const void*)ffb_fused,
                        hipFuncAttributeMaxDynamicSharedMemorySize, SHMEM_BYTES);
    ffb_fused<<<N_PTS / TILE_M, 256, SHMEM_BYTES, stream>>>(
        in_pos, table, ffn_A, W0, b0, bs, bh, wsT, whT, out);
}
